// GCN_71571335020834
// MI455X (gfx1250) — matrix operand order repaired and checked by reading, not compiled, not run on hardware
//
#include <hip/hip_runtime.h>
#include <hip/hip_bf16.h>

// ---------------------------------------------------------------------------
// GCN forward for MI455X (gfx1250): bf16 WMMA GEMMs + f32 atomic scatter.
// ---------------------------------------------------------------------------

#define NN    100000     // nodes
#define NE    1600000    // edges
#define HID   128
#define NG    128        // graphs
#define INDIM 33
#define NCLS  10

typedef __attribute__((ext_vector_type(16))) __bf16 v16bf;
typedef __attribute__((ext_vector_type(8)))  float  v8f;
typedef __attribute__((ext_vector_type(4)))  float  v4f;
typedef __attribute__((ext_vector_type(4)))  __bf16 v4bf;

__device__ __forceinline__ __bf16 f2bf(float f) {
    return __builtin_bit_cast(__bf16, __float2bfloat16(f));
}
__device__ __forceinline__ float bf2f(__bf16 b) {
    return __bfloat162float(__builtin_bit_cast(__hip_bfloat16, b));
}

// ------------------------------ small utilities ----------------------------

__global__ void k_set_f32(float* __restrict__ p, float v, int n) {
    int i = blockIdx.x * blockDim.x + threadIdx.x;
    if (i < n) p[i] = v;
}

__global__ void k_deg(const int* __restrict__ col, const float* __restrict__ w,
                      float* __restrict__ deg, int E) {
    int e = blockIdx.x * blockDim.x + threadIdx.x;
    if (e < E) atomicAdd(&deg[col[e]], w[e]);
}

__global__ void k_dinv(float* __restrict__ deg, int n) {
    int i = blockIdx.x * blockDim.x + threadIdx.x;
    if (i < n) { float d = deg[i]; deg[i] = (d > 0.f) ? rsqrtf(d) : 0.f; }
}

__global__ void k_norm(const int* __restrict__ row, const int* __restrict__ col,
                       const float* __restrict__ w, const float* __restrict__ dinv,
                       float* __restrict__ nrm, int E, int N) {
    int e = blockIdx.x * blockDim.x + threadIdx.x;
    if (e < E) {
        nrm[e] = dinv[row[e]] * w[e] * dinv[col[e]];
    } else if (e < E + N) {
        int i = e - E;
        nrm[e] = dinv[i] * dinv[i];     // self loop, weight 1
    }
}

// --------------------------- embed: h0 = x@We + be -------------------------

__global__ __launch_bounds__(HID)
void k_embed(const float* __restrict__ x, const float* __restrict__ We,
             const float* __restrict__ be, float* __restrict__ h) {
    __shared__ float xs[INDIM];
    int n = blockIdx.x, c = threadIdx.x;
    if (c < INDIM) xs[c] = x[n * INDIM + c];
    __syncthreads();
    float acc = be[c];
#pragma unroll
    for (int k = 0; k < INDIM; ++k) acc = fmaf(xs[k], We[k * HID + c], acc);
    h[n * HID + c] = acc;
}

// ----- weight swizzle: f32 [128x128] -> bf16 B-fragment-major layout -------
// Wswz[((t*4 + s)*32 + lane)*16 + i] = W[k*128 + n],
//   k = s*32 + (i&7) + (i>=8 ? 16 : 0) + (lane>=16 ? 8 : 0),  n = t*16 + (lane&15)
// so each wave's B fragment is a single contiguous, 32B-aligned v16bf load.

__global__ void k_swizzle_w(const float* __restrict__ W, __bf16* __restrict__ Wz) {
    int id = blockIdx.x * blockDim.x + threadIdx.x;
    if (id >= HID * HID) return;
    int i = id & 15;
    int l = (id >> 4) & 31;
    int s = (id >> 9) & 3;
    int t = (id >> 11) & 7;
    int k = s * 32 + (i & 7) + ((i & 8) ? 16 : 0) + ((l & 16) ? 8 : 0);
    int n = t * 16 + (l & 15);
    Wz[id] = f2bf(W[k * HID + n]);
}

// ----------------- GEMM: D(bf16)[N,128] = reluA?(A) @ Wz -------------------
// 256 threads = 8 waves per block; block covers 32 rows x 128 cols.
// Wave w computes the two 16x16 tiles in column-tile w, K=128 in 4 WMMA steps.
// RELU is a compile-time flag so the A-path stays straight-line b128 loads.

template<int RELU>
__global__ __launch_bounds__(256)
void k_gemm_bf16(const float* __restrict__ A, const __bf16* __restrict__ Wz,
                 __bf16* __restrict__ D) {
    const int lane   = threadIdx.x & 31;
    const int wave   = threadIdx.x >> 5;          // column tile 0..7
    const int rowblk = blockIdx.x * 32;
    const int rlo    = lane & 15;
    const int hi     = lane >> 4;                 // 0 or 1

    // A fragment base: row = rowblk + rlo (+16 for second tile),
    // K chunks {kb..kb+7} and {kb+16..kb+23}, kb = ks*32 + 8*hi  (32B aligned)
    const float* base0 = A + (size_t)(rowblk + rlo) * HID + hi * 8;
    const float* base1 = base0 + (size_t)16 * HID;

    v8f c0 = {}; v8f c1 = {};
#pragma unroll
    for (int ks = 0; ks < 4; ++ks) {
        // B fragment: one aligned 32-byte load from swizzled weights
        v16bf b = *(const v16bf*)(Wz + ((((wave * 4 + ks) * 32) + lane) << 4));

        const float* p0 = base0 + ks * 32;
        const float* p1 = base1 + ks * 32;
        v4f q00 = *(const v4f*)(p0);          // k = kb+0..3
        v4f q01 = *(const v4f*)(p0 + 4);      // k = kb+4..7
        v4f q02 = *(const v4f*)(p0 + 16);     // k = kb+16..19
        v4f q03 = *(const v4f*)(p0 + 20);     // k = kb+20..23
        v4f q10 = *(const v4f*)(p1);
        v4f q11 = *(const v4f*)(p1 + 4);
        v4f q12 = *(const v4f*)(p1 + 16);
        v4f q13 = *(const v4f*)(p1 + 20);
        if (RELU) {
#pragma unroll
            for (int i = 0; i < 4; ++i) {
                q00[i] = fmaxf(q00[i], 0.f); q01[i] = fmaxf(q01[i], 0.f);
                q02[i] = fmaxf(q02[i], 0.f); q03[i] = fmaxf(q03[i], 0.f);
                q10[i] = fmaxf(q10[i], 0.f); q11[i] = fmaxf(q11[i], 0.f);
                q12[i] = fmaxf(q12[i], 0.f); q13[i] = fmaxf(q13[i], 0.f);
            }
        }
        v16bf a0, a1;
#pragma unroll
        for (int i = 0; i < 4; ++i) {
            a0[i]      = f2bf(q00[i]);  a0[4 + i]  = f2bf(q01[i]);
            a0[8 + i]  = f2bf(q02[i]);  a0[12 + i] = f2bf(q03[i]);
            a1[i]      = f2bf(q10[i]);  a1[4 + i]  = f2bf(q11[i]);
            a1[8 + i]  = f2bf(q12[i]);  a1[12 + i] = f2bf(q13[i]);
        }
        c0 = __builtin_amdgcn_wmma_f32_16x16x32_bf16(false, a0, false, b,
                                                     (short)0, c0, false, false);
        c1 = __builtin_amdgcn_wmma_f32_16x16x32_bf16(false, a1, false, b,
                                                     (short)0, c1, false, false);
    }
    // C/D layout: element r -> (M = r + 8*hi, N = lane&15)
    const int col = wave * 16 + rlo;
#pragma unroll
    for (int r = 0; r < 8; ++r) {
        int m0 = rowblk + r + hi * 8;
        D[(size_t)m0 * HID + col]        = f2bf(c0[r]);
        D[(size_t)(m0 + 16) * HID + col] = f2bf(c1[r]);
    }
}

// ------------------- seed output with bias, then scatter -------------------

__global__ void k_bias(float* __restrict__ out, const float* __restrict__ b, int total) {
    int i = blockIdx.x * blockDim.x + threadIdx.x;
    if (i < total) out[i] = b[i & (HID - 1)];
}

// out[col2[e]] += norm[e] * msg[row2[e]]   (32 lanes x 4 cols per edge)
__global__ __launch_bounds__(256)
void k_scatter(const int* __restrict__ row, const int* __restrict__ col,
               const float* __restrict__ nrm, const __bf16* __restrict__ msg,
               float* __restrict__ out, int E, int N) {
    int t = blockIdx.x * blockDim.x + threadIdx.x;
    int e = t >> 5;
    if (e >= E + N) return;
    int c4 = (t & 31) * 4;
    int r, d;
    if (e < E) { r = row[e]; d = col[e]; } else { r = e - E; d = r; }
    float nm = nrm[e];
    v4bf m = *(const v4bf*)(msg + (size_t)r * HID + c4);
    float* o = out + (size_t)d * HID + c4;
    atomicAdd(o + 0, nm * bf2f(m[0]));
    atomicAdd(o + 1, nm * bf2f(m[1]));
    atomicAdd(o + 2, nm * bf2f(m[2]));
    atomicAdd(o + 3, nm * bf2f(m[3]));
}

// ------------------------------- pooling + MLP -----------------------------

__global__ __launch_bounds__(HID)
void k_pool(const float* __restrict__ h, const int* __restrict__ batch,
            float* __restrict__ pooled, float* __restrict__ cnt) {
    int n = blockIdx.x, c = threadIdx.x;
    int g = batch[n];
    atomicAdd(&pooled[g * HID + c], fmaxf(h[(size_t)n * HID + c], 0.f));
    if (c == 0) atomicAdd(&cnt[g], 1.0f);
}

__global__ __launch_bounds__(HID)
void k_mlp(const float* __restrict__ pooled, const float* __restrict__ cnt,
           const float* __restrict__ Wr1, const float* __restrict__ br1,
           const float* __restrict__ Wr2, const float* __restrict__ br2,
           const float* __restrict__ Wr3, float* __restrict__ out) {
    __shared__ float p[HID];
    __shared__ float r1[64];
    __shared__ float r2[32];
    int g = blockIdx.x, tid = threadIdx.x;
    float inv = 1.0f / fmaxf(cnt[g], 1.0f);
    p[tid] = pooled[g * HID + tid] * inv;
    __syncthreads();
    if (tid < 64) {
        float a = br1[tid];
        for (int k = 0; k < HID; ++k) a = fmaf(p[k], Wr1[k * 64 + tid], a);
        r1[tid] = fmaxf(a, 0.f);
    }
    __syncthreads();
    if (tid < 32) {
        float a = br2[tid];
        for (int k = 0; k < 64; ++k) a = fmaf(r1[k], Wr2[k * 32 + tid], a);
        r2[tid] = fmaxf(a, 0.f);
    }
    __syncthreads();
    if (tid < NCLS) {
        float a = 0.f;
        for (int k = 0; k < 32; ++k) a = fmaf(r2[k], Wr3[k * NCLS + tid], a);
        out[g * NCLS + tid] = a;
    }
}

// ------------------------------- host driver -------------------------------

static inline int cdiv(long long a, int b) { return (int)((a + b - 1) / b); }

extern "C" void kernel_launch(void* const* d_in, const int* in_sizes, int n_in,
                              void* d_out, int out_size, void* d_ws, size_t ws_size,
                              hipStream_t stream) {
    (void)in_sizes; (void)n_in; (void)out_size; (void)ws_size;

    const float* x   = (const float*)d_in[0];
    const int*   ei  = (const int*)  d_in[1];     // (2, E) row-major
    const float* ew  = (const float*)d_in[2];
    const int*   bat = (const int*)  d_in[3];
    const float* We  = (const float*)d_in[4];
    const float* be  = (const float*)d_in[5];
    const float* W[4]  = { (const float*)d_in[6],  (const float*)d_in[8],
                           (const float*)d_in[10], (const float*)d_in[12] };
    const float* Bv[4] = { (const float*)d_in[7],  (const float*)d_in[9],
                           (const float*)d_in[11], (const float*)d_in[13] };
    const float* Wr1 = (const float*)d_in[14];
    const float* br1 = (const float*)d_in[15];
    const float* Wr2 = (const float*)d_in[16];
    const float* br2 = (const float*)d_in[17];
    const float* Wr3 = (const float*)d_in[18];

    const int* row = ei;            // edge_index[0]
    const int* col = ei + NE;       // edge_index[1]

    // carve workspace (256B aligned chunks)
    char* w = (char*)d_ws;
    auto carve = [&](size_t bytes) {
        char* p = w; w += (bytes + 255) & ~(size_t)255; return p;
    };
    float*  deg    = (float*) carve((size_t)NN * 4);              // -> dinv in place
    float*  nrm    = (float*) carve((size_t)(NE + NN) * 4);
    float*  hA     = (float*) carve((size_t)NN * HID * 4);
    float*  hB     = (float*) carve((size_t)NN * HID * 4);
    __bf16* tmp    = (__bf16*)carve((size_t)NN * HID * 2);        // GEMM out (messages)
    __bf16* Wz     = (__bf16*)carve((size_t)4 * HID * HID * 2);   // swizzled weights
    float*  pooled = (float*) carve((size_t)NG * HID * 4);
    float*  cnt    = (float*) carve((size_t)NG * 4);

    // degree / normalization
    k_set_f32<<<cdiv(NN, 256), 256, 0, stream>>>(deg, 1.0f, NN);  // self-loop weight
    k_deg    <<<cdiv(NE, 256), 256, 0, stream>>>(col, ew, deg, NE);
    k_dinv   <<<cdiv(NN, 256), 256, 0, stream>>>(deg, NN);
    k_norm   <<<cdiv((long long)NE + NN, 256), 256, 0, stream>>>(row, col, ew, deg,
                                                                 nrm, NE, NN);
    // embed
    k_embed<<<NN, HID, 0, stream>>>(x, We, be, hA);

    // weight swizzle (once)
    for (int l = 0; l < 4; ++l)
        k_swizzle_w<<<cdiv(HID * HID, 256), 256, 0, stream>>>(W[l],
                                                              Wz + (size_t)l * HID * HID);
    // 4 GCN layers (ReLU folded into next A-conversion / pooling)
    float* a = hA; float* o = hB;
    for (int l = 0; l < 4; ++l) {
        if (l == 0)
            k_gemm_bf16<0><<<NN / 32, 256, 0, stream>>>(a, Wz, tmp);
        else
            k_gemm_bf16<1><<<NN / 32, 256, 0, stream>>>(a, Wz + (size_t)l * HID * HID,
                                                        tmp);
        k_bias<<<cdiv((long long)NN * HID, 256), 256, 0, stream>>>(o, Bv[l], NN * HID);
        k_scatter<<<cdiv((long long)(NE + NN) * 32, 256), 256, 0, stream>>>(
            row, col, nrm, tmp, o, NE, NN);
        float* t = a; a = o; o = t;
    }
    // after 4 swaps the last conv output is in `a` (== hA)

    // mean pool + MLP head
    k_set_f32<<<cdiv(NG * HID, 256), 256, 0, stream>>>(pooled, 0.f, NG * HID);
    k_set_f32<<<1, 256, 0, stream>>>(cnt, 0.f, NG);
    k_pool<<<NN, HID, 0, stream>>>(a, bat, pooled, cnt);
    k_mlp <<<NG, HID, 0, stream>>>(pooled, cnt, Wr1, br1, Wr2, br2, Wr3,
                                   (float*)d_out);
}
